// GridSampler_65309272703440
// MI455X (gfx1250) — matrix operand order repaired and checked by reading, not compiled, not run on hardware
//
#include <hip/hip_runtime.h>

// GridSampler for MI455X (gfx1250, wave32).
// Hash-based unique (lives in 192MB L2) + tensor-core (WMMA) prefix scans.

#define CELL      0.2f
#define GOFF      64
#define INT_MAXV  0x7fffffff
#define INT_MINV  ((int)0x80000000)

typedef __attribute__((ext_vector_type(2)))  float    v2f;
typedef __attribute__((ext_vector_type(8)))  float    v8f;
typedef __attribute__((ext_vector_type(16))) _Float16 v16h;

// ---------------------------------------------------------------------------
// WMMA 16-element inclusive scan:  D = L * v  with L = 16x16 lower-triangular
// ones.  C/D layout (32-bit, 16x16): VGPR j -> lanes 0-15 hold M=j, lanes
// 16-31 hold M=8+j (N = lane).  So after the matmul, scan[m] sits replicated
// across N in component (m&7), lane-half (m<8 ? lo : hi).
// ---------------------------------------------------------------------------

#if __has_builtin(__builtin_amdgcn_wmma_f32_16x16x4_f32)
// f32 path: K=16 via 4 chained 16x16x4 WMMAs.
// A 16x4 layout: lanes0-15 {K0,K1}, lanes16-31 {K2,K3}; lane&15 = M.
// B 4x16 layout (mirrors C row split): VGPR0 rows {0,2}, VGPR1 rows {1,3}.
__device__ inline v8f wmma_scan16_f32(const float* v16, int lane) {
  const int  mrow = lane & 15;
  const bool hi   = lane >= 16;
  v8f c = {};
#pragma unroll
  for (int j = 0; j < 4; ++j) {
    const int k0 = 4 * j;
    const int kx = k0 + (hi ? 2 : 0);
    const int ky = k0 + (hi ? 3 : 1);
    v2f a, b;
    a.x = (mrow >= kx) ? 1.0f : 0.0f;   // L[m][kx]
    a.y = (mrow >= ky) ? 1.0f : 0.0f;   // L[m][ky]
    b.x = v16[kx];                      // v broadcast across N
    b.y = v16[ky];
    c = __builtin_amdgcn_wmma_f32_16x16x4_f32(false, a, false, b,
                                              (short)0, c, false, false);
  }
  return c;
}
#else
// Fallback: same result layout, scalar math.
__device__ inline v8f wmma_scan16_f32(const float* v16, int lane) {
  const bool hi = lane >= 16;
  v8f c;
#pragma unroll
  for (int j = 0; j < 8; ++j) {
    const int idx = (hi ? 8 : 0) + j;
    float s = 0.0f;
    for (int k = 0; k <= idx; ++k) s += v16[k];
    c[j] = s;
  }
  return c;
}
#endif

// f16 path (probe-verified builtin): single V_WMMA_F32_16X16X32_F16 with a
// lower-triangular f16 A (exact 0/1) and v broadcast in B rows 0..15 (rows
// 16..31 zero).  Exact for integer values <= 2048.
// A 16x32 f16 layout: lanes0-15: elem e -> K = e<8 ? e : e+8;
//                     lanes16-31: elem e -> K = e<8 ? e+8 : e+16.
// B 32x16 f16 layout: same K mapping as A (lanes0-15: e<8 ? e : e+8; lanes16-31: e<8 ? e+8 : e+16).
__device__ inline v8f wmma_scan16_f16(const float* v16, int lane) {
  const int  mrow = lane & 15;
  const bool hi   = lane >= 16;
  v16h a, b;
#pragma unroll
  for (int e = 0; e < 16; ++e) {
    const int ka = hi ? ((e < 8) ? e + 8 : e + 16) : ((e < 8) ? e : e + 8);
    a[e] = (ka < 16 && mrow >= ka) ? (_Float16)1.0f : (_Float16)0.0f;
    const int kb = hi ? ((e < 8) ? e + 8 : e + 16) : ((e < 8) ? e : e + 8);
    b[e] = (kb < 16) ? (_Float16)v16[kb] : (_Float16)0.0f;
  }
  v8f c = {};
  return __builtin_amdgcn_wmma_f32_16x16x32_f16(false, a, false, b,
                                                (short)0, c, false, false);
}

// scan[idx] lives in component (idx&7) on lane 0 (idx<8) or lane 16 (idx>=8).
__device__ inline float scan_at(const float* carr, int idx) {
  return __shfl(carr[idx & 7], (idx < 8) ? 0 : 16, 32);
}

// ---------------------------------------------------------------------------
// Kernels
// ---------------------------------------------------------------------------

__global__ void k_init(int* keys, int* focc, unsigned H, int* stats,
                       int* batch_max) {
  unsigned i      = blockIdx.x * blockDim.x + threadIdx.x;
  unsigned stride = gridDim.x * blockDim.x;
  for (unsigned j = i; j < H; j += stride) {
    keys[j] = -1;
    focc[j] = INT_MAXV;
  }
  if (i == 0) { stats[0] = INT_MAXV; stats[1] = INT_MINV; stats[2] = 0; }
  if (i < 64) batch_max[i] = INT_MINV;
}

__global__ void k_minmax(const float* __restrict__ pts, int n, int* stats) {
  const int i    = blockIdx.x * blockDim.x + threadIdx.x;
  const int lane = threadIdx.x & 31;
  const int wave = threadIdx.x >> 5;
  int lmin = INT_MAXV, lmax = INT_MINV;
  if (i < n) {
    if (i + 4096 < n) __builtin_prefetch(pts + 3 * (i + 4096), 0, 0);
    const int gx = (int)rintf(pts[3 * i + 0] / CELL);
    const int gy = (int)rintf(pts[3 * i + 1] / CELL);
    const int gz = (int)rintf(pts[3 * i + 2] / CELL);
    lmin = min(gx, min(gy, gz));
    lmax = max(gx, max(gy, gz));
  }
#pragma unroll
  for (int o = 16; o > 0; o >>= 1) {
    lmin = min(lmin, __shfl_xor(lmin, o, 32));
    lmax = max(lmax, __shfl_xor(lmax, o, 32));
  }
  __shared__ int smin[8], smax[8];
  if (lane == 0) { smin[wave] = lmin; smax[wave] = lmax; }
  __syncthreads();
  if (threadIdx.x == 0) {
    int m = smin[0], M = smax[0];
    for (int w = 1; w < 8; ++w) { m = min(m, smin[w]); M = max(M, smax[w]); }
    atomicMin(&stats[0], m);
    atomicMax(&stats[1], M);
  }
}

__global__ void k_linidx(const float* __restrict__ pts,
                         const int* __restrict__ bidx, int n,
                         const int* __restrict__ stats, int* __restrict__ lin,
                         int* batch_max) {
  __shared__ int lmax[64];
  if (threadIdx.x < 64) lmax[threadIdx.x] = INT_MINV;
  __syncthreads();
  const int i = blockIdx.x * blockDim.x + threadIdx.x;
  if (i < n) {
    if (i + 4096 < n) __builtin_prefetch(pts + 3 * (i + 4096), 0, 0);
    const int ming = stats[0];
    const int gs   = stats[1] - ming + 2 * GOFF;
    const int gx   = (int)rintf(pts[3 * i + 0] / CELL) - ming + GOFF;
    const int gy   = (int)rintf(pts[3 * i + 1] / CELL) - ming + GOFF;
    const int gz   = (int)rintf(pts[3 * i + 2] / CELL) - ming + GOFF;
    const int li   = gs * (gs * gx + gy) + gz;
    lin[i] = li;
    atomicMax(&lmax[bidx[i] & 63], li);
  }
  __syncthreads();
  if (threadIdx.x < 64 && lmax[threadIdx.x] != INT_MINV)
    atomicMax(&batch_max[threadIdx.x], lmax[threadIdx.x]);
}

__global__ void k_offs(const int* batch_max, const int* bsz_ptr, int* offs,
                       const int* stats, float* out_scalar_gs) {
  if (threadIdx.x == 0) {
    int Bsz = *bsz_ptr;
    if (Bsz > 64) Bsz = 64;
    int cs = 0;
    for (int b = 0; b < Bsz; ++b) {
      offs[b] = cs;
      const int m = batch_max[b];
      if (m != INT_MINV) cs += 2 * m;
    }
    out_scalar_gs[0] = (float)(stats[1] - stats[0] + 2 * GOFF);
  }
}

__global__ void k_hash(const int* __restrict__ bidx, int n, int* __restrict__ lin,
                       const int* __restrict__ offs, int* keys, int* focc,
                       int* __restrict__ slot, unsigned mask) {
  const int i = blockIdx.x * blockDim.x + threadIdx.x;
  if (i >= n) return;
  const int key = lin[i] + offs[bidx[i]];
  lin[i] = key;
  unsigned h = ((unsigned)key * 2654435761u) & mask;
  for (;;) {
    const int prev = atomicCAS(&keys[h], -1, key);
    if (prev == -1 || prev == key) break;
    h = (h + 1) & mask;
  }
  slot[i] = (int)h;
  atomicMin(&focc[h], i);
}

// Per-block count of first-occurrence flags (1024 elems / block).
__global__ void k_bsum(const int* __restrict__ slot,
                       const int* __restrict__ focc, int n,
                       int* __restrict__ bsums) {
  const int base = blockIdx.x * 1024 + threadIdx.x * 4;
  int t = 0;
#pragma unroll
  for (int k = 0; k < 4; ++k) {
    const int i = base + k;
    if (i < n && focc[slot[i]] == i) t++;
  }
#pragma unroll
  for (int o = 16; o > 0; o >>= 1) t += __shfl_xor(t, o, 32);
  __shared__ int s[8];
  if ((threadIdx.x & 31) == 0) s[threadIdx.x >> 5] = t;
  __syncthreads();
  if (threadIdx.x == 0) {
    int tot = 0;
    for (int w = 0; w < 8; ++w) tot += s[w];
    bsums[blockIdx.x] = tot;
  }
}

// Single-block scan of block sums -> exclusive offsets; total = num_cells.
// Cross-wave step uses the f32 tensor-core scan (values up to N).
__global__ void k_scanbsums(int* bsums, int nb, int* stats, float* ncell_out) {
  __shared__ float wsum[16];
  const int tid = threadIdx.x, lane = tid & 31, wave = tid >> 5;
  int carry = 0;
  for (int base = 0; base < nb; base += 4096) {
    int v[16];
    int t = 0;
#pragma unroll
    for (int k = 0; k < 16; ++k) {
      const int idx = base + tid * 16 + k;
      const int x   = (idx < nb) ? bsums[idx] : 0;
      v[k] = x;
      t += x;
    }
    int incl = t;
#pragma unroll
    for (int o = 1; o < 32; o <<= 1) {
      const int y = __shfl_up(incl, o, 32);
      if (lane >= o) incl += y;
    }
    if (tid < 16) wsum[tid] = 0.0f;
    __syncthreads();
    if (lane == 31) wsum[wave] = (float)incl;
    __syncthreads();
    const v8f c = wmma_scan16_f32(wsum, lane);   // EXEC all-ones: uniform call
    float carr[8];
#pragma unroll
    for (int j = 0; j < 8; ++j) carr[j] = c[j];
    const int waveExcl = (wave == 0) ? 0 : (int)scan_at(carr, wave - 1);
    const int total    = (int)scan_at(carr, 7);
    const int texcl    = carry + waveExcl + (incl - t);
    int run = 0;
#pragma unroll
    for (int k = 0; k < 16; ++k) {
      const int idx = base + tid * 16 + k;
      if (idx < nb) bsums[idx] = texcl + run;
      run += v[k];
    }
    carry += total;
    __syncthreads();
  }
  if (tid == 0) { stats[2] = carry; ncell_out[0] = (float)carry; }
}

// Rank first-occurrence points -> cell ids; write per-cell outputs directly.
// Cross-wave step uses the verified f16 WMMA (wave sums <= 128, exact).
__global__ void k_assign(const float* __restrict__ pts,
                         const int* __restrict__ bidx,
                         const int* __restrict__ lin,
                         const int* __restrict__ slot,
                         const int* __restrict__ focc, int n,
                         const int* __restrict__ bsums,
                         int* __restrict__ cellmap,   // aliases keys[]
                         float* __restrict__ out, int N) {
  const int tid = threadIdx.x, lane = tid & 31, wave = tid >> 5;
  const int base = blockIdx.x * 1024 + tid * 4;
  int f[4];
  int t = 0;
#pragma unroll
  for (int k = 0; k < 4; ++k) {
    const int i = base + k;
    f[k] = (i < n && focc[slot[i]] == i) ? 1 : 0;
    t += f[k];
  }
  int incl = t;
#pragma unroll
  for (int o = 1; o < 32; o <<= 1) {
    const int y = __shfl_up(incl, o, 32);
    if (lane >= o) incl += y;
  }
  __shared__ float wsum[16];
  if (tid < 16) wsum[tid] = 0.0f;
  __syncthreads();
  if (lane == 31) wsum[wave] = (float)incl;
  __syncthreads();
  const v8f c = wmma_scan16_f16(wsum, lane);     // EXEC all-ones: uniform call
  float carr[8];
#pragma unroll
  for (int j = 0; j < 8; ++j) carr[j] = c[j];
  const int waveExcl = (wave == 0) ? 0 : (int)scan_at(carr, wave - 1);
  int rank = bsums[blockIdx.x] + waveExcl + (incl - t);
#pragma unroll
  for (int k = 0; k < 4; ++k) {
    const int i = base + k;
    if (f[k]) {
      const int cidx = rank;
      cellmap[slot[i]]   = cidx;
      out[cidx]          = (float)bidx[i];     // batch_idx_out
      out[N + cidx]      = (float)lin[i];      // unique_linear_idx
      const size_t ro = (size_t)3 * N + 3 * (size_t)cidx;
      out[ro + 0] = CELL * rintf(pts[3 * i + 0] / CELL);
      out[ro + 1] = CELL * rintf(pts[3 * i + 1] / CELL);
      out[ro + 2] = CELL * rintf(pts[3 * i + 2] / CELL);
    }
    rank += f[k];
  }
}

// Per-point inverse map + padding of invalid cells.
__global__ void k_cellout(const int* __restrict__ slot,
                          const int* __restrict__ cellmap,
                          const int* __restrict__ stats, int n,
                          float* __restrict__ out, int N) {
  const int i = blockIdx.x * blockDim.x + threadIdx.x;
  if (i >= n) return;
  out[(size_t)2 * N + i] = (float)cellmap[slot[i]];
  if (i >= stats[2]) {
    out[i]     = -1.0f;
    out[N + i] = -1.0f;
    const size_t ro = (size_t)3 * N + 3 * (size_t)i;
    out[ro + 0] = 0.0f; out[ro + 1] = 0.0f; out[ro + 2] = 0.0f;
  }
}

// ---------------------------------------------------------------------------

extern "C" void kernel_launch(void* const* d_in, const int* in_sizes, int n_in,
                              void* d_out, int out_size, void* d_ws,
                              size_t ws_size, hipStream_t stream) {
  const int    N    = in_sizes[0] / 3;
  const float* pts  = (const float*)d_in[0];
  const int*   bidx = (const int*)d_in[1];
  const int*   bsz  = (const int*)d_in[2];
  float*       out  = (float*)d_out;

  unsigned H = 1024;
  while (H < 2u * (unsigned)N) H <<= 1;   // >= 2x points: probe always ends

  char* p = (char*)d_ws;
  auto take = [&](size_t bytes) {
    char* q = p;
    p += (bytes + 255) & ~(size_t)255;
    return q;
  };
  int* stats     = (int*)take(16 * sizeof(int));
  int* batch_max = (int*)take(64 * sizeof(int));
  int* offs      = (int*)take(65 * sizeof(int));
  int* lin       = (int*)take((size_t)N * sizeof(int));
  int* slot      = (int*)take((size_t)N * sizeof(int));
  int* keys      = (int*)take((size_t)H * sizeof(int));  // later: cellmap
  int* focc      = (int*)take((size_t)H * sizeof(int));
  const int nb   = (N + 1023) / 1024;
  int* bsums     = (int*)take((size_t)nb * sizeof(int));
  (void)ws_size; (void)n_in; (void)out_size;

  const int gp = (N + 255) / 256;

  k_init<<<2048, 256, 0, stream>>>(keys, focc, H, stats, batch_max);
  k_minmax<<<gp, 256, 0, stream>>>(pts, N, stats);
  k_linidx<<<gp, 256, 0, stream>>>(pts, bidx, N, stats, lin, batch_max);
  k_offs<<<1, 32, 0, stream>>>(batch_max, bsz, offs, stats,
                               out + (size_t)6 * N);
  k_hash<<<gp, 256, 0, stream>>>(bidx, N, lin, offs, keys, focc, slot, H - 1);
  k_bsum<<<nb, 256, 0, stream>>>(slot, focc, N, bsums);
  k_scanbsums<<<1, 256, 0, stream>>>(bsums, nb, stats,
                                     out + (size_t)6 * N + 1);
  k_assign<<<nb, 256, 0, stream>>>(pts, bidx, lin, slot, focc, N, bsums,
                                   keys /*cellmap*/, out, N);
  k_cellout<<<gp, 256, 0, stream>>>(slot, keys, stats, N, out, N);
}
